// LinearNVFP4_24713241821312
// MI455X (gfx1250) — matrix operand order repaired and checked by reading, not compiled, not run on hardware
//
#include <hip/hip_runtime.h>

// ---------------------------------------------------------------------------
// LinearNVFP4: out = x @ fake_quant_nvfp4(weight)^T + bias
//   x: [M=32768, K=2048] f32, weight: [N=2048, K=2048] f32, bias: [N] f32
// Plan: (1) fake-quant weight -> bf16 (exact: fp4 * E4M3 has <= 6 sig bits)
//       (2) bf16 WMMA GEMM (v_wmma_f32_16x16x32_bf16), f32 accumulate
//           - A tile (x): global f32 -> regs -> cvt bf16 -> LDS (all 8 waves)
//           - B tile (wq): Tensor Data Mover DMA global -> LDS with native
//             padding (row stride 72 halves), waited on TENSORcnt
// ---------------------------------------------------------------------------

typedef __bf16 v16bf __attribute__((ext_vector_type(16)));
typedef __bf16 v8bf  __attribute__((ext_vector_type(8)));
typedef float  v8f   __attribute__((ext_vector_type(8)));
typedef float  v4f   __attribute__((ext_vector_type(4)));

typedef unsigned int u32x4 __attribute__((ext_vector_type(4)));
typedef int          i32x8 __attribute__((ext_vector_type(8)));
typedef int          i32x4 __attribute__((ext_vector_type(4)));

__device__ __forceinline__ unsigned short f2bf_rne(float f) {
  unsigned int u = __float_as_uint(f);
  unsigned int r = 0x7FFFu + ((u >> 16) & 1u);   // round-to-nearest-even
  u += r;
  return (unsigned short)(u >> 16);
}

// ---------------- Kernel 1: NVFP4 fake-quant of the weight ------------------
__global__ void nvfp4_quant_weight(const float* __restrict__ w,
                                   unsigned short* __restrict__ wq,  // bf16 bits
                                   int nblocks) {
  int b = blockIdx.x * blockDim.x + threadIdx.x;
  if (b >= nblocks) return;
  const float* p = w + (size_t)b * 16;

  float v[16];
  float amax = 0.f;
#pragma unroll
  for (int i = 0; i < 16; ++i) {
    v[i] = p[i];
    amax = fmaxf(amax, fabsf(v[i]));
  }
  float scale = fmaxf(amax * (1.0f / 6.0f), 1e-7f);

  // E4M3 encode (trunc toward zero) -> decode
  int e = (int)truncf(log2f(scale)) + 7;
  e = min(max(e, 0), 15);
  float p2   = exp2f((float)(e - 7));
  float mv   = scale / p2;
  float mant = (mv >= 1.0f) ? fminf(truncf((mv - 1.0f) * 8.0f), 7.0f) : 0.0f;
  float sdec = (1.0f + mant * 0.125f) * p2;
  if (e == 0 && mant == 0.0f) sdec = 0.0f;

  const float tbl[8] = {0.f, 0.5f, 1.f, 1.5f, 2.f, 3.f, 4.f, 6.f};
  union { unsigned short s[16]; uint4 q[2]; } out;
#pragma unroll
  for (int i = 0; i < 16; ++i) {
    float nrm = rintf(v[i] / scale);     // RNE, matches jnp.round
    float sgn = (nrm > 0.f) ? 1.f : ((nrm < 0.f) ? -1.f : 0.f);
    float mag = fabsf(nrm);
    float best = 3.4e38f, q = 0.f;
#pragma unroll
    for (int j = 0; j < 8; ++j) {        // nearest E2M1, first-on-tie
      float d = fabsf(mag - tbl[j]);
      if (d < best) { best = d; q = tbl[j]; }
    }
    out.s[i] = f2bf_rne(sgn * q * sdec); // exact in bf16
  }
  uint4* dst = (uint4*)(wq + (size_t)b * 16);
  dst[0] = out.q[0];
  dst[1] = out.q[1];
}

// ---------------- Kernel 2: bf16 WMMA GEMM with TDM B-tiles ----------------
#define BM 128
#define BN 128
#define BK 64
#define LDT 72   // LDS row stride in halves: 144B = 9*16B (aligned, bank-conflict-free)

__global__ __launch_bounds__(256)
void gemm_bf16_wmma(const float* __restrict__ X,            // [M,K] f32
                    const unsigned short* __restrict__ Wq,  // [N,K] bf16 bits
                    const float* __restrict__ bias,         // [N]
                    float* __restrict__ Out,                // [M,N] f32
                    int M, int N, int K) {
  __shared__ unsigned short lA[BM * LDT];
  __shared__ unsigned short lB[BN * LDT];

  const int tid  = threadIdx.x;
  const int wave = tid >> 5;
  const int lane = tid & 31;
  const int lh   = lane >> 4;   // lane half
  const int l16  = lane & 15;

  const int m0 = blockIdx.y * BM;
  const int n0 = blockIdx.x * BN;

  // 8 waves arranged 2(M) x 4(N): each wave computes 64x32 = 4x2 tiles of 16x16
  const int wm0 = (wave >> 2) * 64;
  const int wn0 = (wave & 3) * 32;

  // Invariant parts of the TDM descriptor (group1: sizes/strides/padding).
  // data_size=2B; pad_enable; pad_interval=4 (32 DWORDs = 128B = tile row);
  // pad_amount=3 (4 DWORDs = 16B) -> LDS row stride 144B = LDT halves.
  const unsigned ldsB = (unsigned)(size_t)(void*)lB;    // LDS byte offset
  i32x8 g1;
  g1[0] = (1 << 16) | (1 << 20) | (4 << 22) | (3 << 25);
  g1[1] = (K & 0xFFFF) << 16;                           // tensor_dim0 lo
  g1[2] = ((K >> 16) & 0xFFFF) | ((N & 0xFFFF) << 16);  // dim0 hi | dim1 lo
  g1[3] = ((N >> 16) & 0xFFFF) | (BK << 16);            // dim1 hi | tile_dim0
  g1[4] = BN;                                           // tile_dim1 (rows)
  g1[5] = K;                                            // tensor_dim0_stride
  g1[6] = 0;
  g1[7] = 0;
  const i32x4 gz4 = {0, 0, 0, 0};                       // groups 2/3: 2-D tensor
  const i32x8 gz8 = {0, 0, 0, 0, 0, 0, 0, 0};

  v8f acc[4][2];
#pragma unroll
  for (int i = 0; i < 4; ++i)
#pragma unroll
    for (int j = 0; j < 2; ++j)
      acc[i][j] = (v8f){0.f, 0.f, 0.f, 0.f, 0.f, 0.f, 0.f, 0.f};

  for (int kt = 0; kt < K; kt += BK) {
    // ---- B tile via Tensor Data Mover: Wq[n0..n0+127][kt..kt+63] -> lB ----
    if (wave == 0) {
      unsigned long long ga =
          (unsigned long long)(size_t)Wq + ((size_t)n0 * K + kt) * 2ull;
      u32x4 g0;
      g0.x = 1u;                                   // count=1 (valid D#)
      g0.y = ldsB;                                 // lds_addr
      g0.z = (unsigned)ga;                         // global_addr lo
      g0.w = (unsigned)(ga >> 32) | 0x80000000u;   // global_addr hi | type=2
      __builtin_amdgcn_tensor_load_to_lds(g0, g1, gz4, gz4, gz8, 0);
    }

    // ---- A tile: x[m0..m0+127][kt..kt+63] f32 -> bf16 LDS (all waves) ----
#pragma unroll
    for (int i = 0; i < 8; ++i) {
      int idx = i * 256 + tid;                 // float4 index; 16 per row
      int row = idx >> 4;
      int c4  = idx & 15;
      const float* src = X + (size_t)(m0 + row) * K + kt + c4 * 4;
      if (kt + BK < K) __builtin_prefetch(src + BK, 0, 1);  // global_prefetch_b8
      v4f xv = *(const v4f*)src;
      unsigned int p0 = (unsigned)f2bf_rne(xv.x) | ((unsigned)f2bf_rne(xv.y) << 16);
      unsigned int p1 = (unsigned)f2bf_rne(xv.z) | ((unsigned)f2bf_rne(xv.w) << 16);
      *(uint2*)&lA[row * LDT + c4 * 4] = make_uint2(p0, p1);
    }

    if (wave == 0) __builtin_amdgcn_s_wait_tensorcnt(0);  // TDM done -> LDS valid
    __syncthreads();

#pragma unroll
    for (int kk = 0; kk < BK; kk += 32) {
      // A fragment (16x32 bf16): lane<16 holds K 0-15; lane>=16 holds K 16-31
      v16bf a[4];
#pragma unroll
      for (int i = 0; i < 4; ++i) {
        const unsigned short* base = &lA[(wm0 + i * 16 + l16) * LDT + kk + lh * 16];
        v8bf lo = *(const v8bf*)(base);        // 16B aligned
        v8bf hi = *(const v8bf*)(base + 8);
        a[i] = __builtin_shufflevector(lo, hi, 0, 1, 2, 3, 4, 5, 6, 7,
                                               8, 9, 10, 11, 12, 13, 14, 15);
      }
      // B fragment (32x16 bf16): lane holds column N=l16, contiguous K 0-15 / 16-31
      v16bf b[2];
#pragma unroll
      for (int j = 0; j < 2; ++j) {
        const unsigned short* base = &lB[(wn0 + j * 16 + l16) * LDT + kk + lh * 16];
        v8bf lo = *(const v8bf*)(base);
        v8bf hi = *(const v8bf*)(base + 8);
        b[j] = __builtin_shufflevector(lo, hi, 0, 1, 2, 3, 4, 5, 6, 7,
                                               8, 9, 10, 11, 12, 13, 14, 15);
      }
#pragma unroll
      for (int i = 0; i < 4; ++i)
#pragma unroll
        for (int j = 0; j < 2; ++j)
          acc[i][j] = __builtin_amdgcn_wmma_f32_16x16x32_bf16(
              false, a[i], false, b[j], (short)0, acc[i][j], false, false);
    }
    __syncthreads();
  }

  // ---- epilogue: C layout -> VGPR r holds (M = r | 8+r, N = lane&15) ----
#pragma unroll
  for (int j = 0; j < 2; ++j) {
    int n = n0 + wn0 + j * 16 + l16;
    float bv = bias[n];
#pragma unroll
    for (int i = 0; i < 4; ++i) {
      int mbase = m0 + wm0 + i * 16 + lh * 8;
#pragma unroll
      for (int r = 0; r < 8; ++r)
        Out[(size_t)(mbase + r) * N + n] = acc[i][j][r] + bv;
    }
  }
}

// ---------------------------------------------------------------------------
extern "C" void kernel_launch(void* const* d_in, const int* in_sizes, int n_in,
                              void* d_out, int out_size, void* d_ws, size_t ws_size,
                              hipStream_t stream) {
  const float* x    = (const float*)d_in[0];
  const float* w    = (const float*)d_in[1];
  const float* bias = (const float*)d_in[2];
  float* out        = (float*)d_out;

  const int N = in_sizes[2];                  // 2048
  const int K = in_sizes[1] / N;              // 2048
  const int M = (int)((long)in_sizes[0] / K); // 32768

  unsigned short* wq = (unsigned short*)d_ws; // bf16 bits, N*K*2 bytes (8 MB)

  const int nblk = in_sizes[1] / 16;
  nvfp4_quant_weight<<<(nblk + 255) / 256, 256, 0, stream>>>(w, wq, nblk);

  dim3 grid(N / BN, M / BM);                  // (16, 256)
  gemm_bf16_wmma<<<grid, 256, 0, stream>>>(x, wq, bias, out, M, N, K);
}
